// GNNAnomalyDetector_91070486544698
// MI455X (gfx1250) — matrix operand order repaired and checked by reading, not compiled, not run on hardware
//
#include <hip/hip_runtime.h>
#include <hip/hip_fp16.h>

typedef __attribute__((ext_vector_type(16))) _Float16 v16h;
typedef __attribute__((ext_vector_type(8)))  float    v8f;

union Frag { v16h h; uint32_t u[8]; uint4 q[2]; };

#define THREADS 256

// ---------------- prep kernels ----------------

// Pack f32 weight [K][Nn] into TRANSPOSED u32 f16-pair layout [Nn][K/2]:
// out[n*(K/2) + kp] = {W[2kp][n], W[2kp+1][n]}  -> B fragments become contiguous b128s.
__global__ void k_pack_weight_T(const float* __restrict__ W, uint32_t* __restrict__ out,
                                int K, int Nn) {
  int idx = blockIdx.x * blockDim.x + threadIdx.x;
  int KP = K >> 1;
  int total = KP * Nn;
  if (idx >= total) return;
  int n = idx / KP, kp = idx - n * KP;
  __half lo = __float2half(W[(2 * kp) * Nn + n]);
  __half hi = __float2half(W[(2 * kp + 1) * Nn + n]);
  out[idx] = (uint32_t)__half_as_ushort(lo) | ((uint32_t)__half_as_ushort(hi) << 16);
}

// Fold bias+BN: s = g*rsqrt(v+eps); t = (b-m)*s + be
__global__ void k_scale_shift(const float* __restrict__ g, const float* __restrict__ be,
                              const float* __restrict__ m, const float* __restrict__ v,
                              const float* __restrict__ b,
                              float* __restrict__ s, float* __restrict__ t, int C) {
  int c = blockIdx.x * blockDim.x + threadIdx.x;
  if (c >= C) return;
  float sc = g[c] * rsqrtf(v[c] + 1e-5f);
  s[c] = sc;
  t[c] = (b[c] - m[c]) * sc + be[c];
}

__global__ void k_f32_to_h(const float* __restrict__ in, __half* __restrict__ out, int n) {
  int i = blockIdx.x * blockDim.x + threadIdx.x;
  if (i < n) out[i] = __float2half(in[i]);
}

__global__ void k_deg_init(float* __restrict__ deg, int n) {
  int i = blockIdx.x * blockDim.x + threadIdx.x;
  if (i < n) deg[i] = 1.0f;                       // self-loop
}

__global__ void k_deg_acc(const int* __restrict__ dst, float* __restrict__ deg, int E) {
  int e = blockIdx.x * blockDim.x + threadIdx.x;
  if (e < E) unsafeAtomicAdd(&deg[dst[e]], 1.0f);
}

__global__ void k_deg_inv(const float* __restrict__ deg, float* __restrict__ dinv, int n) {
  int i = blockIdx.x * blockDim.x + threadIdx.x;
  if (i < n) dinv[i] = rsqrtf(deg[i]);
}

// agg[i][c] = hlin[i][c] * dinv[i]^2  (self-loop term = initializer, C/4 float4 per node)
__global__ void k_self_init(const float* __restrict__ hlin, const float* __restrict__ dinv,
                            float* __restrict__ agg, int n, int C4) {
  int idx = blockIdx.x * blockDim.x + threadIdx.x;
  if (idx >= n * C4) return;
  int i = idx / C4, c4 = idx - i * C4;
  float w = dinv[i] * dinv[i];
  float4 v = ((const float4*)hlin)[(size_t)i * C4 + c4];
  v.x *= w; v.y *= w; v.z *= w; v.w *= w;
  ((float4*)agg)[(size_t)i * C4 + c4] = v;
}

// agg[dst] += hlin[src] * dinv[src]*dinv[dst]  (one thread = 4 channels of one edge)
__global__ void k_edge_scatter(const int* __restrict__ src, const int* __restrict__ dst,
                               const float* __restrict__ dinv, const float* __restrict__ hlin,
                               float* __restrict__ agg, int E, int C4) {
  int idx = blockIdx.x * blockDim.x + threadIdx.x;
  if (idx >= E * C4) return;
  int e = idx / C4, c4 = idx - e * C4;
  int s = src[e], d = dst[e];
  float nrm = dinv[s] * dinv[d];
  float4 v = ((const float4*)hlin)[(size_t)s * C4 + c4];
  float* ap = agg + (size_t)d * (C4 * 4) + c4 * 4;
  unsafeAtomicAdd(ap + 0, v.x * nrm);
  unsafeAtomicAdd(ap + 1, v.y * nrm);
  unsafeAtomicAdd(ap + 2, v.z * nrm);
  unsafeAtomicAdd(ap + 3, v.w * nrm);
}

// out[i] = f16(relu(agg[i]*s[c] + t[c]))  (C is power of 2)
__global__ void k_bn_relu_h(const float* __restrict__ agg, const float* __restrict__ s,
                            const float* __restrict__ t, __half* __restrict__ out,
                            int total, int cmask) {
  int i = blockIdx.x * blockDim.x + threadIdx.x;
  if (i >= total) return;
  int c = i & cmask;
  float y = fmaxf(agg[i] * s[c] + t[c], 0.0f);
  out[i] = __float2half(y);
}

// out[e] = sigmoid(dot(z3[e][0:64], Wc4) + bc4)
__global__ void k_final(const __half* __restrict__ z3, const float* __restrict__ Wc4,
                        const float* __restrict__ bc4, float* __restrict__ out, int E) {
  int e = blockIdx.x * blockDim.x + threadIdx.x;
  if (e >= E) return;
  const __half* row = z3 + (size_t)e * 64;
  float acc = 0.f;
#pragma unroll
  for (int c = 0; c < 64; ++c) acc += __half2float(row[c]) * Wc4[c];
  acc += bc4[0];
  out[e] = 1.0f / (1.0f + __expf(-acc));
}

// ---------------- WMMA GEMM ----------------
// C[M][Nn] = A[M][K] @ B[K][Nn], f16 in, f32 accum via v_wmma_f32_16x16x32_f16.
// Block = 128 rows, 8 waves; wave w owns rows [w*16, w*16+16), loops all NT n-tiles.
// K is chunked 64-wide through LDS. LDS rows padded to 36 dwords: 16B-aligned
// (b128 fragment loads) and bank-scrambled (36 = 4*9, 9 coprime 16).
// B tile is held TRANSPOSED in LDS ([n][kp]) so each B fragment = 2x ds_load_b128,
// and is staged with global_load_async_to_lds_b128 (ASYNCcnt DMA, no VGPR bounce).
// Fragment loads are software-pipelined one wmma ahead to avoid dscnt==0 stalls.
// EDGE: A row = concat(h2h[src[e]], h2h[dst[e]]) gathered during staging; the 2
// edge_attr features are folded into the epilogue as a rank-2 update.
template<int K, int NT, bool EDGE>
__global__ __launch_bounds__(256) void k_gemm(
    const __half* __restrict__ A,
    const int* __restrict__ srcIdx, const int* __restrict__ dstIdx,
    const uint32_t* __restrict__ h2u,                 // [N][128] u32 pairs (EDGE)
    const uint32_t* __restrict__ BpT,                 // [Nn][K/2] packed pairs (transposed)
    const float* __restrict__ scale, const float* __restrict__ shift,
    const float* __restrict__ ea,                     // [M][3] (EDGE)
    const float* __restrict__ eaW0, const float* __restrict__ eaW1,
    float* __restrict__ outF, __half* __restrict__ outH,
    int M, int relu)
{
  constexpr int Nn  = NT * 16;
  constexpr int KP  = K / 2;     // u32 pairs per row
  constexpr int KP4 = KP / 4;    // uint4 per row
  constexpr int KCP = 32;        // pairs per K-chunk (64 f16)
  constexpr int CHUNKS = KP / KCP;
  constexpr int LS = KCP + 4;    // LDS row stride (36 dwords = 144B, 16B aligned)
  __shared__ alignas(16) uint32_t sA[128 * LS];
  __shared__ alignas(16) uint32_t sB[Nn * LS];

  const int tid   = threadIdx.x;
  const int lane  = tid & 31;
  const int wv    = tid >> 5;      // wave id -> 16-row strip
  const int half_ = lane >> 4;
  const int m16   = lane & 15;
  const int r0    = blockIdx.x * 128;

  const uint4* Au4  = (const uint4*)A;
  const uint4* h2u4 = (const uint4*)h2u;   // row stride 32 uint4
  const uint4* Bp4  = (const uint4*)BpT;   // row stride KP4

  v8f acc[NT];
  v8f zero = {0.f, 0.f, 0.f, 0.f, 0.f, 0.f, 0.f, 0.f};
#pragma unroll
  for (int nt = 0; nt < NT; ++nt) acc[nt] = zero;

  auto loadA = [&](Frag& a, int kl) {
    const uint32_t* arow = &sA[(wv * 16 + m16) * LS + kl * 16];
    a.q[0] = *(const uint4*)(arow + 4 * half_);       // K = 8*half + 0..7
    a.q[1] = *(const uint4*)(arow + 8 + 4 * half_);   // K = 16 + 8*half + 0..7
  };
  auto loadB = [&](Frag& b, int kl, int nt) {
    const uint32_t* bbase = &sB[(nt * 16 + m16) * LS + kl * 16];
    b.q[0] = *(const uint4*)(bbase + 4 * half_);      // K = 8*half + 0..7
    b.q[1] = *(const uint4*)(bbase + 8 + 4 * half_);  // K = 16 + 8*half + 0..7
  };

  for (int kc = 0; kc < CHUNKS; ++kc) {
    if (kc) __syncthreads();
    // stage A chunk: 128 rows x 8 uint4 (gathered for EDGE), guarded + zero-filled
#pragma unroll
    for (int it = 0; it < (128 * 8) / THREADS; ++it) {
      int idx = it * THREADS + tid;
      int r = idx >> 3;
      int c4 = idx & 7;
      int row = r0 + r;
      uint4 val = make_uint4(0u, 0u, 0u, 0u);
      if (row < M) {
        if (EDGE) {
          int cg4 = kc * 8 + c4;                   // uint4 col in [0,64)
          int node = (cg4 < 32) ? srcIdx[row] : dstIdx[row];
          int cc   = (cg4 < 32) ? cg4 : cg4 - 32;
          val = h2u4[(size_t)node * 32 + cc];
        } else {
          val = Au4[(size_t)row * KP4 + kc * 8 + c4];
          if (kc + 1 < CHUNKS)                      // prefetch next chunk of A stream
            __builtin_prefetch(&Au4[(size_t)row * KP4 + (kc + 1) * 8 + c4], 0, 1);
        }
      }
      *(uint4*)&sA[r * LS + c4 * 4] = val;
    }
    // stage B chunk: Nn rows x 8 uint4 via async DMA to LDS (no guards needed)
#pragma unroll
    for (int it = 0; it < (Nn * 8) / THREADS; ++it) {
      int idx = it * THREADS + tid;
      int n = idx >> 3;
      int kp4 = idx & 7;
      unsigned ldsAddr = (unsigned)(size_t)(&sB[n * LS + kp4 * 4]);
      unsigned long long gAddr =
          (unsigned long long)(size_t)(&Bp4[(size_t)n * KP4 + kc * 8 + kp4]);
      asm volatile("global_load_async_to_lds_b128 %0, %1, off"
                   :: "v"(ldsAddr), "v"(gAddr) : "memory");
    }
    asm volatile("s_wait_asynccnt 0x0" ::: "memory");
    __syncthreads();

#pragma unroll
    for (int kl = 0; kl < 2; ++kl) {           // two K=32 steps per chunk
      Frag a;
      loadA(a, kl);
      Frag bf[2];
      loadB(bf[0], kl, 0);
#pragma unroll
      for (int nt = 0; nt < NT; ++nt) {
        if (nt + 1 < NT) loadB(bf[(nt + 1) & 1], kl, nt + 1);   // pipeline 1 ahead
        acc[nt] = __builtin_amdgcn_wmma_f32_16x16x32_f16(
            false, a.h, false, bf[nt & 1].h, (short)0, acc[nt], false, false);
      }
    }
  }

  // epilogue: (+edge_attr rank-2) * scale + shift, relu, store f32/f16
#pragma unroll
  for (int nt = 0; nt < NT; ++nt) {
    int col = nt * 16 + m16;
    float s = scale ? scale[col] : 1.0f;
    float t = shift ? shift[col] : 0.0f;
    float w0 = 0.f, w1 = 0.f;
    if (EDGE) { w0 = eaW0[col]; w1 = eaW1[col]; }
#pragma unroll
    for (int v = 0; v < 8; ++v) {
      int row = r0 + wv * 16 + 8 * half_ + v;
      if (row < M) {
        float y = acc[nt][v];
        if (EDGE) y += ea[(size_t)row * 3 + 0] * w0 + ea[(size_t)row * 3 + 1] * w1;
        y = y * s + t;
        if (relu) y = fmaxf(y, 0.0f);
        if (outF) outF[(size_t)row * Nn + col] = y;
        if (outH) outH[(size_t)row * Nn + col] = __float2half(y);
      }
    }
  }
}

// ---------------- host launch ----------------
extern "C" void kernel_launch(void* const* d_in, const int* in_sizes, int n_in,
                              void* d_out, int out_size, void* d_ws, size_t ws_size,
                              hipStream_t stream) {
  (void)n_in; (void)out_size; (void)ws_size;
  const float* x   = (const float*)d_in[0];
  const int*   ei  = (const int*)d_in[1];
  const float* ea  = (const float*)d_in[2];
  const float* W1  = (const float*)d_in[3];
  const float* b1  = (const float*)d_in[4];
  const float* g1  = (const float*)d_in[5];
  const float* be1 = (const float*)d_in[6];
  const float* m1  = (const float*)d_in[7];
  const float* v1  = (const float*)d_in[8];
  const float* W2  = (const float*)d_in[9];
  const float* b2  = (const float*)d_in[10];
  const float* g2  = (const float*)d_in[11];
  const float* be2 = (const float*)d_in[12];
  const float* m2  = (const float*)d_in[13];
  const float* v2  = (const float*)d_in[14];
  const float* Wc1 = (const float*)d_in[15];
  const float* bc1 = (const float*)d_in[16];
  const float* gc1 = (const float*)d_in[17];
  const float* bec1= (const float*)d_in[18];
  const float* mc1 = (const float*)d_in[19];
  const float* vc1 = (const float*)d_in[20];
  const float* Wc2 = (const float*)d_in[21];
  const float* bc2 = (const float*)d_in[22];
  const float* gc2 = (const float*)d_in[23];
  const float* bec2= (const float*)d_in[24];
  const float* mc2 = (const float*)d_in[25];
  const float* vc2 = (const float*)d_in[26];
  const float* Wc3 = (const float*)d_in[27];
  const float* bc3 = (const float*)d_in[28];
  const float* Wc4 = (const float*)d_in[29];
  const float* bc4 = (const float*)d_in[30];

  const int N = in_sizes[0] / 128;
  const int E = in_sizes[1] / 2;
  const int* src = ei;
  const int* dst = ei + E;

  char* ws = (char*)d_ws;
  size_t cur = 0;
  auto alloc = [&](size_t bytes) { size_t off = cur; cur += (bytes + 255) & ~(size_t)255; return off; };

  size_t oW1p  = alloc(64 * 128 * 4);
  size_t oW2p  = alloc(64 * 256 * 4);
  size_t oWc1p = alloc(256 * 256 * 4);
  size_t oWc2p = alloc(128 * 128 * 4);
  size_t oWc3p = alloc(64 * 64 * 4);
  size_t oS1 = alloc(128 * 4), oT1 = alloc(128 * 4);
  size_t oS2 = alloc(256 * 4), oT2 = alloc(256 * 4);
  size_t oSc1 = alloc(256 * 4), oTc1 = alloc(256 * 4);
  size_t oSc2 = alloc(128 * 4), oTc2 = alloc(128 * 4);
  size_t oDeg  = alloc((size_t)N * 4);
  size_t oDinv = alloc((size_t)N * 4);
  size_t oXh    = alloc((size_t)N * 128 * 2);
  size_t oH1lin = alloc((size_t)N * 128 * 4);
  size_t oAgg1  = alloc((size_t)N * 128 * 4);
  size_t oH1h   = alloc((size_t)N * 128 * 2);
  size_t oH2lin = alloc((size_t)N * 256 * 4);
  size_t oAgg2  = alloc((size_t)N * 256 * 4);
  size_t oH2h   = alloc((size_t)N * 256 * 2);
  size_t oZ1    = alloc((size_t)E * 256 * 2);
  // z2/z3 reuse the dead region [oDeg .. oZ1): (deg..h2h dead once z1 exists)
  size_t oZ2 = oDeg;
  size_t oZ3 = oDeg + (((size_t)E * 128 * 2 + 255) & ~(size_t)255);

  uint32_t* W1p  = (uint32_t*)(ws + oW1p);
  uint32_t* W2p  = (uint32_t*)(ws + oW2p);
  uint32_t* Wc1p = (uint32_t*)(ws + oWc1p);
  uint32_t* Wc2p = (uint32_t*)(ws + oWc2p);
  uint32_t* Wc3p = (uint32_t*)(ws + oWc3p);
  float *s1 = (float*)(ws + oS1), *t1 = (float*)(ws + oT1);
  float *s2 = (float*)(ws + oS2), *t2 = (float*)(ws + oT2);
  float *sc1 = (float*)(ws + oSc1), *tc1 = (float*)(ws + oTc1);
  float *sc2 = (float*)(ws + oSc2), *tc2 = (float*)(ws + oTc2);
  float *deg = (float*)(ws + oDeg), *dinv = (float*)(ws + oDinv);
  __half* xh    = (__half*)(ws + oXh);
  float*  h1lin = (float*)(ws + oH1lin);
  float*  agg1  = (float*)(ws + oAgg1);
  __half* h1h   = (__half*)(ws + oH1h);
  float*  h2lin = (float*)(ws + oH2lin);
  float*  agg2  = (float*)(ws + oAgg2);
  __half* h2h   = (__half*)(ws + oH2h);
  __half* z1    = (__half*)(ws + oZ1);
  __half* z2    = (__half*)(ws + oZ2);
  __half* z3    = (__half*)(ws + oZ3);

  auto cdiv = [](long long a, long long b) { return (int)((a + b - 1) / b); };
  dim3 blk(THREADS);

  // --- prep ---
  k_pack_weight_T<<<cdiv(64 * 128, THREADS), blk, 0, stream>>>(W1, W1p, 128, 128);
  k_pack_weight_T<<<cdiv(64 * 256, THREADS), blk, 0, stream>>>(W2, W2p, 128, 256);
  k_pack_weight_T<<<cdiv(256 * 256, THREADS), blk, 0, stream>>>(Wc1, Wc1p, 512, 256);
  k_pack_weight_T<<<cdiv(128 * 128, THREADS), blk, 0, stream>>>(Wc2, Wc2p, 256, 128);
  k_pack_weight_T<<<cdiv(64 * 64, THREADS), blk, 0, stream>>>(Wc3, Wc3p, 128, 64);
  k_scale_shift<<<1, blk, 0, stream>>>(g1, be1, m1, v1, b1, s1, t1, 128);
  k_scale_shift<<<1, blk, 0, stream>>>(g2, be2, m2, v2, b2, s2, t2, 256);
  k_scale_shift<<<1, blk, 0, stream>>>(gc1, bec1, mc1, vc1, bc1, sc1, tc1, 256);
  k_scale_shift<<<1, blk, 0, stream>>>(gc2, bec2, mc2, vc2, bc2, sc2, tc2, 128);
  k_f32_to_h<<<cdiv((long long)N * 128, THREADS), blk, 0, stream>>>(x, xh, N * 128);

  // --- degrees ---
  k_deg_init<<<cdiv(N, THREADS), blk, 0, stream>>>(deg, N);
  k_deg_acc<<<cdiv(E, THREADS), blk, 0, stream>>>(dst, deg, E);
  k_deg_inv<<<cdiv(N, THREADS), blk, 0, stream>>>(deg, dinv, N);

  // --- GCN layer 1: h1 = relu(BN(scatter(x@W1) + b1)) ---
  k_gemm<128, 8, false><<<cdiv(N, 128), blk, 0, stream>>>(
      xh, nullptr, nullptr, nullptr, W1p, nullptr, nullptr,
      nullptr, nullptr, nullptr, h1lin, nullptr, N, 0);
  k_self_init<<<cdiv((long long)N * 32, THREADS), blk, 0, stream>>>(h1lin, dinv, agg1, N, 32);
  k_edge_scatter<<<cdiv((long long)E * 32, THREADS), blk, 0, stream>>>(src, dst, dinv, h1lin, agg1, E, 32);
  k_bn_relu_h<<<cdiv((long long)N * 128, THREADS), blk, 0, stream>>>(agg1, s1, t1, h1h, N * 128, 127);

  // --- GCN layer 2: h2 = relu(BN(scatter(h1@W2) + b2)), kept in f16 for gathers ---
  k_gemm<128, 16, false><<<cdiv(N, 128), blk, 0, stream>>>(
      h1h, nullptr, nullptr, nullptr, W2p, nullptr, nullptr,
      nullptr, nullptr, nullptr, h2lin, nullptr, N, 0);
  k_self_init<<<cdiv((long long)N * 64, THREADS), blk, 0, stream>>>(h2lin, dinv, agg2, N, 64);
  k_edge_scatter<<<cdiv((long long)E * 64, THREADS), blk, 0, stream>>>(src, dst, dinv, h2lin, agg2, E, 64);
  k_bn_relu_h<<<cdiv((long long)N * 256, THREADS), blk, 0, stream>>>(agg2, s2, t2, h2h, N * 256, 255);

  // --- edge MLP layer 1: z1 = relu(BN(concat(h2[src],h2[dst],ea2)@Wc1 + bc1)) (gather fused) ---
  k_gemm<512, 16, true><<<cdiv(E, 128), blk, 0, stream>>>(
      nullptr, src, dst, (const uint32_t*)h2h, Wc1p, sc1, tc1,
      ea, Wc1 + 512 * 256, Wc1 + 513 * 256, nullptr, z1, E, 1);

  // --- edge MLP layer 2: z2 = relu(BN(z1@Wc2 + bc2)) ---
  k_gemm<256, 8, false><<<cdiv(E, 128), blk, 0, stream>>>(
      z1, nullptr, nullptr, nullptr, Wc2p, sc2, tc2,
      nullptr, nullptr, nullptr, nullptr, z2, E, 1);

  // --- edge MLP layer 3: z3 = relu(z2@Wc3 + bc3) ---
  k_gemm<128, 4, false><<<cdiv(E, 128), blk, 0, stream>>>(
      z2, nullptr, nullptr, nullptr, Wc3p, nullptr, bc3,
      nullptr, nullptr, nullptr, nullptr, z3, E, 1);

  // --- output: sigmoid(z3@Wc4 + bc4) ---
  k_final<<<cdiv(E, THREADS), blk, 0, stream>>>(z3, Wc4, bc4, (float*)d_out, E);
}
